// PVConv_11149735100554
// MI455X (gfx1250) — matrix operand order repaired and checked by reading, not compiled, not run on hardware
//
#include <hip/hip_runtime.h>
#include <stdint.h>

#define BDIM 8
#define CDIM 64
#define NPTS 32768
#define OCD  64
#define RDIM 32
#define R3   32768
#define EPSV   1e-6f
#define BN_EPS 1e-4f
#define ALPHA  0.1f

typedef __attribute__((ext_vector_type(16))) __bf16 v16bf;
typedef __attribute__((ext_vector_type(8)))  float  v8f;

union ABu { v16bf v; uint4 q[2]; };

__device__ __forceinline__ unsigned short f2bf(float f) {
  union { float f; uint32_t u; } x; x.f = f;
  uint32_t r = x.u + 0x7FFFu + ((x.u >> 16) & 1u);   // round-to-nearest-even
  return (unsigned short)(r >> 16);
}
__device__ __forceinline__ float lrelu(float x) { return x >= 0.f ? x : ALPHA * x; }

// ---------------- zero scratch (cnt + vox sums) ----------------
__global__ void k_zero(float4* p, int n4) {
  int i = blockIdx.x * blockDim.x + threadIdx.x;
  if (i < n4) p[i] = make_float4(0.f, 0.f, 0.f, 0.f);
}

// ---------------- fold BN+bias into per-channel affine; cast wp ----------------
__global__ void k_prep(const float* b1, const float* g1, const float* be1, const float* m1, const float* v1,
                       const float* b2, const float* g2, const float* be2, const float* m2, const float* v2,
                       const float* bp, const float* gp, const float* bep, const float* mp, const float* vp,
                       const float* wp, float* bn, unsigned short* wpb) {
  int t = threadIdx.x;
  if (t < 64) {
    float s = g1[t] * rsqrtf(v1[t] + BN_EPS);
    bn[t] = s; bn[64 + t] = (b1[t] - m1[t]) * s + be1[t];
  } else if (t < 128) {
    int o = t - 64;
    float s = g2[o] * rsqrtf(v2[o] + BN_EPS);
    bn[128 + o] = s; bn[192 + o] = (b2[o] - m2[o]) * s + be2[o];
  } else if (t < 192) {
    int o = t - 128;
    float s = gp[o] * rsqrtf(vp[o] + BN_EPS);
    bn[256 + o] = s; bn[320 + o] = (bp[o] - mp[o]) * s + bep[o];
  }
  for (int i = t; i < OCD * CDIM; i += 256) wpb[i] = f2bf(wp[i]);
}

// ---------------- conv weights (OC,C,27) f32 -> (27,OC,C) bf16 ----------------
__global__ void k_wcast(const float* w1, const float* w2, unsigned short* wb1, unsigned short* wb2) {
  int i = blockIdx.x * blockDim.x + threadIdx.x;
  if (i >= 2 * OCD * CDIM * 27) return;
  const float* src = (i < OCD * CDIM * 27) ? w1 : w2;
  unsigned short* dst = (i < OCD * CDIM * 27) ? wb1 : wb2;
  int j = (i < OCD * CDIM * 27) ? i : i - OCD * CDIM * 27;
  int oc = j / (CDIM * 27);
  int rem = j % (CDIM * 27);
  int c = rem / 27, ko = rem % 27;
  dst[(ko * OCD + oc) * CDIM + c] = f2bf(src[j]);
}

// ---------------- per-batch coord mean + max-norm -> scale ----------------
__global__ void k_stats(const float* coords, float* stats) {
  __shared__ float red[256];
  __shared__ float mean[3];
  int b = blockIdx.x, t = threadIdx.x;
  const float* cb = coords + (size_t)b * 3 * NPTS;
  for (int d = 0; d < 3; ++d) {
    float s = 0.f;
    for (int n = t; n < NPTS; n += 256) s += cb[(size_t)d * NPTS + n];
    red[t] = s; __syncthreads();
    for (int o = 128; o > 0; o >>= 1) { if (t < o) red[t] += red[t + o]; __syncthreads(); }
    if (t == 0) mean[d] = red[0] * (1.0f / NPTS);
    __syncthreads();
  }
  float m0 = mean[0], m1 = mean[1], m2 = mean[2], mx = 0.f;
  for (int n = t; n < NPTS; n += 256) {
    float x = cb[n] - m0, y = cb[NPTS + n] - m1, z = cb[2 * NPTS + n] - m2;
    mx = fmaxf(mx, sqrtf(x * x + y * y + z * z));
  }
  red[t] = mx; __syncthreads();
  for (int o = 128; o > 0; o >>= 1) { if (t < o) red[t] = fmaxf(red[t], red[t + o]); __syncthreads(); }
  if (t == 0) {
    stats[b * 4 + 0] = m0; stats[b * 4 + 1] = m1; stats[b * 4 + 2] = m2;
    stats[b * 4 + 3] = 1.0f / (2.0f * red[0] + EPSV);
  }
}

// ---------------- voxelize: float coords + atomic scatter of features ----------------
__global__ void k_voxelize(const float* __restrict__ coords, const float* __restrict__ features,
                           const float* __restrict__ stats, float* nc, float* cnt, float* vsum) {
  int gid = blockIdx.x * blockDim.x + threadIdx.x;
  if (gid >= BDIM * NPTS) return;
  int b = gid >> 15, n = gid & (NPTS - 1);
  const float* st = stats + b * 4;
  float s = st[3];
  int vi[3];
#pragma unroll
  for (int d = 0; d < 3; ++d) {
    float c = coords[((size_t)b * 3 + d) * NPTS + n] - st[d];
    float v = fminf(fmaxf((c * s + 0.5f) * (float)RDIM, 0.0f), (float)(RDIM - 1));
    nc[((size_t)b * 3 + d) * NPTS + n] = v;
    vi[d] = (int)rintf(v);
  }
  int flat = (vi[0] * RDIM + vi[1]) * RDIM + vi[2];
  atomicAdd(cnt + (size_t)b * R3 + flat, 1.0f);
  float* dst = vsum + (((size_t)b * R3 + flat) << 6);
  const float* src = features + ((size_t)b * CDIM) * NPTS + n;
#pragma unroll 8
  for (int c = 0; c < CDIM; ++c) atomicAdd(dst + c, src[(size_t)c * NPTS]);
}

// ---------------- mean-normalize voxels, cast to bf16 (B,R3,C) ----------------
__global__ void k_normcast(const float* __restrict__ vsum, const float* __restrict__ cnt,
                           unsigned short* vbf) {
  int i = blockIdx.x * blockDim.x + threadIdx.x;
  if (i >= BDIM * R3 * CDIM) return;
  float c = cnt[i >> 6];
  vbf[i] = f2bf(vsum[i] / fmaxf(c, 1.0f));
}

// ---------------- features (B,C,N) f32 -> (B,N,C) bf16 ----------------
__global__ void k_featcast(const float* __restrict__ feat, unsigned short* fbf) {
  int i = blockIdx.x * blockDim.x + threadIdx.x;
  if (i >= BDIM * NPTS * CDIM) return;
  int b = i >> 21;
  int r = i & ((1 << 21) - 1);
  int n = r >> 6, c = r & 63;
  fbf[i] = f2bf(feat[(((size_t)b << 6) + c) * NPTS + n]);
}

// ---------------- implicit-GEMM 3x3x3 conv + BN + LeakyReLU (WMMA bf16) ----------------
// 4 waves/block; wave w -> oc tile [16w,16w+16) over 32 consecutive voxels (full z-row):
// two 16x16 accumulators per wave, A (weights) loaded once per K-step, used twice.
__global__ __launch_bounds__(128) void k_conv(const unsigned short* __restrict__ xin_u,
                                              const unsigned short* __restrict__ w_u,
                                              const float* __restrict__ bnA, const float* __restrict__ bnB,
                                              unsigned short* out_bf, float* out_f) {
  const __bf16* xin = (const __bf16*)xin_u;
  const __bf16* w   = (const __bf16*)w_u;
  int wave = threadIdx.x >> 5;
  int lane = threadIdx.x & 31;
  int half = lane >> 4, lm = lane & 15;
  int b  = blockIdx.x >> 10;              // 1024 tiles of 32 voxels per batch
  int s0 = (blockIdx.x & 1023) << 5;      // z-aligned: s0 % 32 == 0
  int ocb = wave << 4;
  int x = s0 >> 10, y = (s0 >> 5) & 31;
  int z0 = lm, z1 = lm + 16;

  v8f acc0 = {0.f, 0.f, 0.f, 0.f, 0.f, 0.f, 0.f, 0.f};
  v8f acc1 = {0.f, 0.f, 0.f, 0.f, 0.f, 0.f, 0.f, 0.f};
  const __bf16* wrow = w + ((size_t)(ocb + lm) << 6);
  size_t xbase = ((size_t)b << 15);

  for (int dx = -1; dx <= 1; ++dx) {
    int nx = x + dx;
    if ((unsigned)nx >= 32u) continue;            // uniform: zero-pad tap plane skipped
    for (int dy = -1; dy <= 1; ++dy) {
      int ny = y + dy;
      if ((unsigned)ny >= 32u) continue;          // uniform
      const __bf16* xrow = xin + ((xbase + (size_t)(((nx << 5) + ny) << 5)) << 6);
#pragma unroll
      for (int dz = -1; dz <= 1; ++dz) {
        int ko = (dx + 1) * 9 + (dy + 1) * 3 + (dz + 1);
        const __bf16* wk = wrow + ((size_t)ko << 12);   // ko*64*64
        int nz0 = z0 + dz, nz1 = z1 + dz;
        bool v0 = (unsigned)nz0 < 32u;                  // per-lane (z edges only)
        bool v1 = (unsigned)nz1 < 32u;
        const __bf16* xb0 = xrow + ((size_t)nz0 << 6);
        const __bf16* xb1 = xrow + ((size_t)nz1 << 6);
        __builtin_prefetch(wk + 4096, 0, 3);
#pragma unroll
        for (int cc = 0; cc < 2; ++cc) {
          int cb = cc << 5;
          ABu A, B0, B1;
          // A (16x32 bf16): lane-half picks contiguous K run of 16
          A.q[0] = *(const uint4*)(wk + cb + (half << 4));
          A.q[1] = *(const uint4*)(wk + cb + (half << 4) + 8);
          // B (32x16 bf16): lane-half picks contiguous K run of 16
          if (v0) {
            const __bf16* p = xb0 + cb + (half << 4);
            B0.q[0] = *(const uint4*)(p);
            B0.q[1] = *(const uint4*)(p + 8);
          } else {
            B0.q[0] = make_uint4(0u, 0u, 0u, 0u);
            B0.q[1] = make_uint4(0u, 0u, 0u, 0u);
          }
          if (v1) {
            const __bf16* p = xb1 + cb + (half << 4);
            B1.q[0] = *(const uint4*)(p);
            B1.q[1] = *(const uint4*)(p + 8);
          } else {
            B1.q[0] = make_uint4(0u, 0u, 0u, 0u);
            B1.q[1] = make_uint4(0u, 0u, 0u, 0u);
          }
          acc0 = __builtin_amdgcn_wmma_f32_16x16x32_bf16(false, A.v, false, B0.v,
                                                         (short)0, acc0, false, false);
          acc1 = __builtin_amdgcn_wmma_f32_16x16x32_bf16(false, A.v, false, B1.v,
                                                         (short)0, acc1, false, false);
        }
      }
    }
  }

  int p0 = s0 + lm, p1 = s0 + 16 + lm;
  size_t ob0 = (xbase + (size_t)p0) << 6;
  size_t ob1 = (xbase + (size_t)p1) << 6;
#pragma unroll
  for (int j = 0; j < 8; ++j) {
    int oc = ocb + j + (half << 3);
    float t0 = lrelu(acc0[j] * bnA[oc] + bnB[oc]);
    float t1 = lrelu(acc1[j] * bnA[oc] + bnB[oc]);
    if (out_bf) { out_bf[ob0 + oc] = f2bf(t0); out_bf[ob1 + oc] = f2bf(t1); }
    if (out_f)  { out_f[ob0 + oc]  = t0;       out_f[ob1 + oc]  = t1; }
  }
}

// ---------------- trilinear devoxelize: writes fused = devox into d_out ----------------
__global__ void k_devox(const float* __restrict__ vox, const float* __restrict__ nc, float* out) {
  int gid = blockIdx.x * blockDim.x + threadIdx.x;    // B*N*64 threads
  if (gid >= BDIM * NPTS * CDIM) return;
  int c = gid & 63;
  int pid = gid >> 6;
  int b = pid >> 15, n = pid & (NPTS - 1);
  float fx = nc[((size_t)b * 3 + 0) * NPTS + n];
  float fy = nc[((size_t)b * 3 + 1) * NPTS + n];
  float fz = nc[((size_t)b * 3 + 2) * NPTS + n];
  float flx = floorf(fx), fly = floorf(fy), flz = floorf(fz);
  float dx = fx - flx, dy = fy - fly, dz = fz - flz;
  int x0 = (int)flx, y0 = (int)fly, z0 = (int)flz;
  int x1 = min(x0 + 1, RDIM - 1), y1 = min(y0 + 1, RDIM - 1), z1 = min(z0 + 1, RDIM - 1);
  const float* vb = vox + (((size_t)b << 15) << 6) + c;
  float acc = 0.f;
#pragma unroll
  for (int k = 0; k < 8; ++k) {
    int ix = (k & 4) ? x1 : x0;
    int iy = (k & 2) ? y1 : y0;
    int iz = (k & 1) ? z1 : z0;
    float wg = ((k & 4) ? dx : 1.f - dx) * ((k & 2) ? dy : 1.f - dy) * ((k & 1) ? dz : 1.f - dz);
    size_t vidx = (size_t)((((ix << 5) + iy) << 5) + iz) << 6;
    acc += wg * vb[vidx];
  }
  out[(((size_t)b << 6) + c) * NPTS + n] = acc;
}

// ---------------- pointwise 64x64 GEMM + BN + LeakyReLU, fused add (WMMA bf16) -------
// wave -> 16 oc x 32 points, two accumulators sharing each A fragment
__global__ __launch_bounds__(128) void k_point(const unsigned short* __restrict__ fbf_u,
                                               const unsigned short* __restrict__ wpb_u,
                                               const float* __restrict__ bnA,
                                               const float* __restrict__ bnB, float* out) {
  const __bf16* fbf = (const __bf16*)fbf_u;
  const __bf16* wpb = (const __bf16*)wpb_u;
  int wave = threadIdx.x >> 5;
  int lane = threadIdx.x & 31;
  int half = lane >> 4, lm = lane & 15;
  int b  = blockIdx.x >> 10;              // 1024 tiles of 32 points per batch
  int n0 = (blockIdx.x & 1023) << 5;
  int ocb = wave << 4;
  int na = n0 + lm, nb = n0 + 16 + lm;
  const __bf16* xb0 = fbf + (((size_t)b * NPTS + (size_t)na) << 6);
  const __bf16* xb1 = fbf + (((size_t)b * NPTS + (size_t)nb) << 6);
  const __bf16* wr  = wpb + ((size_t)(ocb + lm) << 6);

  v8f acc0 = {0.f, 0.f, 0.f, 0.f, 0.f, 0.f, 0.f, 0.f};
  v8f acc1 = {0.f, 0.f, 0.f, 0.f, 0.f, 0.f, 0.f, 0.f};
#pragma unroll
  for (int cc = 0; cc < 2; ++cc) {
    int cb = cc << 5;
    ABu A, B0, B1;
    A.q[0]  = *(const uint4*)(wr + cb + (half << 4));
    A.q[1]  = *(const uint4*)(wr + cb + (half << 4) + 8);
    B0.q[0] = *(const uint4*)(xb0 + cb + (half << 4));
    B0.q[1] = *(const uint4*)(xb0 + cb + (half << 4) + 8);
    B1.q[0] = *(const uint4*)(xb1 + cb + (half << 4));
    B1.q[1] = *(const uint4*)(xb1 + cb + (half << 4) + 8);
    acc0 = __builtin_amdgcn_wmma_f32_16x16x32_bf16(false, A.v, false, B0.v,
                                                   (short)0, acc0, false, false);
    acc1 = __builtin_amdgcn_wmma_f32_16x16x32_bf16(false, A.v, false, B1.v,
                                                   (short)0, acc1, false, false);
  }
#pragma unroll
  for (int j = 0; j < 8; ++j) {
    int oc = ocb + j + (half << 3);
    float t0 = lrelu(acc0[j] * bnA[oc] + bnB[oc]);
    float t1 = lrelu(acc1[j] * bnA[oc] + bnB[oc]);
    size_t o0 = (((size_t)b << 6) + (size_t)oc) * NPTS + (size_t)na;
    size_t o1 = (((size_t)b << 6) + (size_t)oc) * NPTS + (size_t)nb;
    out[o0] += t0;                // devox wrote first; tiles disjoint -> race-free
    out[o1] += t1;
  }
}

extern "C" void kernel_launch(void* const* d_in, const int* in_sizes, int n_in,
                              void* d_out, int out_size, void* d_ws, size_t ws_size,
                              hipStream_t stream) {
  (void)in_sizes; (void)n_in; (void)out_size; (void)ws_size;
  const float* features = (const float*)d_in[0];
  const float* coords   = (const float*)d_in[1];
  const float* w1  = (const float*)d_in[2];
  const float* b1  = (const float*)d_in[3];
  const float* g1  = (const float*)d_in[4];
  const float* be1 = (const float*)d_in[5];
  const float* m1  = (const float*)d_in[6];
  const float* v1  = (const float*)d_in[7];
  const float* w2  = (const float*)d_in[8];
  const float* b2  = (const float*)d_in[9];
  const float* g2  = (const float*)d_in[10];
  const float* be2 = (const float*)d_in[11];
  const float* m2  = (const float*)d_in[12];
  const float* v2  = (const float*)d_in[13];
  const float* wp  = (const float*)d_in[14];
  const float* bp  = (const float*)d_in[15];
  const float* gp  = (const float*)d_in[16];
  const float* bep = (const float*)d_in[17];
  const float* mp  = (const float*)d_in[18];
  const float* vp  = (const float*)d_in[19];

  char* ws = (char*)d_ws;
  float* stats          = (float*)(ws + 0);                        // 8*4 f32
  float* bn             = (float*)(ws + 256);                      // 384 f32
  float* ncb            = (float*)(ws + 2048);                     // B*3*N f32   = 3 MiB
  float* cnt            = (float*)(ws + 2048 + 3145728);           // B*R3 f32    = 1 MiB
  float* vsum           = (float*)(ws + 2048 + 3145728 + 1048576); // B*R3*64 f32 = 64 MiB (reused as conv2 f32 out)
  unsigned short* vbf   = (unsigned short*)((char*)vsum + 67108864); // 32 MiB
  unsigned short* c1bf  = (unsigned short*)((char*)vbf  + 33554432); // 32 MiB
  unsigned short* wb1   = (unsigned short*)((char*)c1bf + 33554432); // 27*64*64 bf16
  unsigned short* wb2   = (unsigned short*)((char*)wb1  + 262144);
  unsigned short* wpb   = (unsigned short*)((char*)wb2  + 262144);
  unsigned short* fbf   = (unsigned short*)((char*)wpb  + 8192);     // 32 MiB

  float* outF = (float*)d_out;

  int zn4 = (1048576 + 67108864) / 16;
  k_zero<<<(zn4 + 255) / 256, 256, 0, stream>>>((float4*)cnt, zn4);
  k_prep<<<1, 256, 0, stream>>>(b1, g1, be1, m1, v1, b2, g2, be2, m2, v2,
                                bp, gp, bep, mp, vp, wp, bn, wpb);
  k_wcast<<<(2 * OCD * CDIM * 27 + 255) / 256, 256, 0, stream>>>(w1, w2, wb1, wb2);
  k_stats<<<BDIM, 256, 0, stream>>>(coords, stats);
  k_voxelize<<<(BDIM * NPTS) / 256, 256, 0, stream>>>(coords, features, stats, ncb, cnt, vsum);
  k_normcast<<<(BDIM * R3 * CDIM) / 256, 256, 0, stream>>>(vsum, cnt, vbf);
  k_featcast<<<(BDIM * NPTS * CDIM) / 256, 256, 0, stream>>>(features, fbf);
  k_conv<<<BDIM * (R3 / 32), 128, 0, stream>>>(vbf, wb1, bn + 0, bn + 64, c1bf, nullptr);
  k_conv<<<BDIM * (R3 / 32), 128, 0, stream>>>(c1bf, wb2, bn + 128, bn + 192, nullptr, vsum);
  k_devox<<<(BDIM * NPTS * CDIM) / 256, 256, 0, stream>>>(vsum, ncb, outF);
  k_point<<<BDIM * (NPTS / 32), 128, 0, stream>>>(fbf, wpb, bn + 256, bn + 320, outF);
  hipMemcpyAsync(outF + (size_t)BDIM * OCD * NPTS, coords,
                 (size_t)BDIM * 3 * NPTS * sizeof(float), hipMemcpyDeviceToDevice, stream);
}
